// Network_18485539242046
// MI455X (gfx1250) — matrix operand order repaired and checked by reading, not compiled, not run on hardware
//
#include <hip/hip_runtime.h>
#include <hip/hip_bf16.h>

// ---------------------------------------------------------------------------
// Problem constants (from reference): B=64, L=10, D=768, DIS=50, K=9, SEM=128
// ---------------------------------------------------------------------------
#define D_DIM    768
#define DIS_DIM  50
#define DP_DIM   1586            // 2*D + DIS
#define DP_PAD   1600            // padded to multiple of 32 for WMMA K
#define D2_DIM   1536            // 2*D
#define SEM_DIM  128
#define NSENT    640             // B*L
#define NPAIR    6400            // B*L*L

// ---------------------------------------------------------------------------
// WMMA types / helpers (CDNA5 wave32, V_WMMA_F32_16X16X32_BF16)
// ---------------------------------------------------------------------------
typedef __attribute__((ext_vector_type(16))) __bf16        v16bf;
typedef __attribute__((ext_vector_type(8)))  float         v8f;
typedef __attribute__((ext_vector_type(4)))  unsigned int  u32x4;

union BFFrag { v16bf v; u32x4 q[2]; };   // 32 bytes = 16 bf16 = one WMMA operand

__device__ __forceinline__ unsigned short f2bf(float f) {
  unsigned u = __float_as_uint(f);
  return (unsigned short)((u + 0x7fffu + ((u >> 16) & 1u)) >> 16);   // RNE
}
__device__ __forceinline__ float bf2f(unsigned short s) {
  return __uint_as_float(((unsigned)s) << 16);
}

// ---------------------------------------------------------------------------
// GEMM on pre-split bf16 operands:
//   C[M,N] = act( (Ahi+Alo)[M,Kp] @ (Wthi+Wtlo)^T[N,Kp]^T + bias )
// A buffers: row-major [M x Kp] bf16 (hi/lo).  W buffers: TRANSPOSED
// row-major [N x Kp] bf16 (hi/lo), K zero-padded.  Split-bf16 product:
// hi*hi + hi*lo + lo*hi, f32 accumulation (~tf32-or-better accuracy).
// Block = 128 thr = 4 waves; wave w -> rows [64bx+16w,+16), cols [64by,+64).
// Requires M%64==0, N%64==0, Kp%32==0. act: 0=none, 1=relu.
// ---------------------------------------------------------------------------
__global__ __launch_bounds__(128)
void wmma_gemm_bf(const unsigned short* __restrict__ Ahi,
                  const unsigned short* __restrict__ Alo,
                  const unsigned short* __restrict__ Wthi,
                  const unsigned short* __restrict__ Wtlo,
                  const float* __restrict__ bias, float* __restrict__ C,
                  int N, int Kp, int act) {
  const int lane = threadIdx.x & 31;
  const int wave = threadIdx.x >> 5;
  const int hl   = lane >> 4;          // lane half: 0 => lanes 0-15, 1 => 16-31
  const int l16  = lane & 15;
  const int mBase = blockIdx.x * 64 + wave * 16;
  const int nBase = blockIdx.y * 64;

  v8f acc[4] = {};

  // A fragment: lane row m=mBase+l16; K slices {0..7}U{16..23} (+8 if hl)
  const unsigned short* aH = Ahi + (size_t)(mBase + l16) * Kp + hl * 8;
  const unsigned short* aL = Alo + (size_t)(mBase + l16) * Kp + hl * 8;
  // B fragment: lane row n (Wt is N-major); K slices {0..7}U{16..23} (+8 if hl)
  const unsigned short* bH0 = Wthi + (size_t)(nBase + l16) * Kp + hl * 8;
  const unsigned short* bL0 = Wtlo + (size_t)(nBase + l16) * Kp + hl * 8;
  const size_t tStride = (size_t)16 * Kp;   // next 16-wide N tile

  for (int kb = 0; kb < Kp; kb += 32) {
    BFFrag ah, al;
    ah.q[0] = *(const u32x4*)(aH + kb);
    ah.q[1] = *(const u32x4*)(aH + kb + 16);
    al.q[0] = *(const u32x4*)(aL + kb);
    al.q[1] = *(const u32x4*)(aL + kb + 16);
#pragma unroll
    for (int t = 0; t < 4; ++t) {
      BFFrag bh, bl;
      const unsigned short* ph = bH0 + t * tStride + kb;
      const unsigned short* pl = bL0 + t * tStride + kb;
      bh.q[0] = *(const u32x4*)(ph);
      bh.q[1] = *(const u32x4*)(ph + 16);
      bl.q[0] = *(const u32x4*)(pl);
      bl.q[1] = *(const u32x4*)(pl + 16);
      acc[t] = __builtin_amdgcn_wmma_f32_16x16x32_bf16(false, ah.v, false, bh.v,
                                                       (short)0, acc[t], false, false);
      acc[t] = __builtin_amdgcn_wmma_f32_16x16x32_bf16(false, ah.v, false, bl.v,
                                                       (short)0, acc[t], false, false);
      acc[t] = __builtin_amdgcn_wmma_f32_16x16x32_bf16(false, al.v, false, bh.v,
                                                       (short)0, acc[t], false, false);
    }
  }

  // epilogue: C/D layout (lane -> N=l16, vgpr r -> M = r + 8*hl)
#pragma unroll
  for (int t = 0; t < 4; ++t) {
    const int n = nBase + t * 16 + l16;
    const float bv = bias ? bias[n] : 0.0f;
#pragma unroll
    for (int r = 0; r < 8; ++r) {
      const int m = mBase + r + hl * 8;
      float v = acc[t][r] + bv;
      if (act == 1) v = fmaxf(v, 0.0f);
      C[(size_t)m * N + n] = v;
    }
  }
}

// ---------------------------------------------------------------------------
// Split f32 [M x K] row-major -> bf16 hi/lo [M x Kp] (zero-pad K..Kp)
// ---------------------------------------------------------------------------
__global__ void split_act(const float* __restrict__ X, unsigned short* __restrict__ hi,
                          unsigned short* __restrict__ lo, int M, int K, int Kp) {
  size_t idx = blockIdx.x * (size_t)blockDim.x + threadIdx.x;
  size_t total = (size_t)M * Kp;
  if (idx >= total) return;
  int k = (int)(idx % Kp);
  size_t m = idx / Kp;
  float x = (k < K) ? X[m * K + k] : 0.0f;
  unsigned short h = f2bf(x);
  hi[idx] = h;
  lo[idx] = f2bf(x - bf2f(h));
}

// Weight f32 [K x N] -> transposed split bf16 [N x Kp] (zero-pad K..Kp)
__global__ void split_weightT(const float* __restrict__ W, unsigned short* __restrict__ hi,
                              unsigned short* __restrict__ lo, int K, int N, int Kp) {
  size_t idx = blockIdx.x * (size_t)blockDim.x + threadIdx.x;
  size_t total = (size_t)N * Kp;
  if (idx >= total) return;
  int k = (int)(idx % Kp);
  int n = (int)(idx / Kp);
  float x = (k < K) ? W[(size_t)k * N + n] : 0.0f;
  unsigned short h = f2bf(x);
  hi[idx] = h;
  lo[idx] = f2bf(x - bf2f(h));
}

// ---------------------------------------------------------------------------
// Gather-concat -> split bf16 [NPAIR x DP_PAD]:
// row p = [xa[ia[p]], xb[ib[p]], dist_table[ptag[p]], 0pad]
// ---------------------------------------------------------------------------
__global__ void build_concat_split(const float* __restrict__ xa, const float* __restrict__ xb,
                                   const float* __restrict__ dt,
                                   const int* __restrict__ ia, const int* __restrict__ ib,
                                   const int* __restrict__ ptag,
                                   unsigned short* __restrict__ hi,
                                   unsigned short* __restrict__ lo) {
  const int p = blockIdx.x;
  const float* ra = xa + (size_t)ia[p] * D_DIM;
  const float* rb = xb + (size_t)ib[p] * D_DIM;
  const float* rd = dt + (size_t)ptag[p] * DIS_DIM;
  const size_t base = (size_t)p * DP_PAD;
  for (int c = threadIdx.x; c < DP_PAD; c += blockDim.x) {
    float v;
    if (c < D_DIM)            v = ra[c];
    else if (c < 2 * D_DIM)   v = rb[c - D_DIM];
    else if (c < DP_DIM)      v = rd[c - 2 * D_DIM];
    else                      v = 0.0f;
    unsigned short h = f2bf(v);
    hi[base + c] = h;
    lo[base + c] = f2bf(v - bf2f(h));
  }
}

// out[r] = dot(X[r,:], w) + bias (one wave32 per row)
__global__ void rowdot(const float* __restrict__ X, const float* __restrict__ w,
                       const float* __restrict__ bias, float* __restrict__ out,
                       int rows, int K) {
  const int wv = (blockIdx.x * blockDim.x + threadIdx.x) >> 5;
  const int lane = threadIdx.x & 31;
  if (wv >= rows) return;
  const float* row = X + (size_t)wv * K;
  float s = 0.0f;
  for (int k = lane; k < K; k += 32) s += row[k] * w[k];
#pragma unroll
  for (int off = 16; off > 0; off >>= 1) s += __shfl_down(s, off, 32);
  if (lane == 0) out[wv] = s + (bias ? bias[0] : 0.0f);
}

// ---------------------------------------------------------------------------
// GAT aggregate for statically-regular graphs: dst d has 10 in-neighbors at
// rows base + i*stride, base = (d/divA)*mulA + (d%modB)*mulB.
// e_i = leaky_relu(a_src[nbr]+a_dst[d], 0.2); softmax; out = elu(sum alpha*z).
// ---------------------------------------------------------------------------
__global__ void gat_agg(const float* __restrict__ zsrc, const float* __restrict__ as,
                        const float* __restrict__ ad, float* __restrict__ out,
                        int F, int divA, int mulA, int modB, int mulB, int stride) {
  const int d = blockIdx.x;
  const int base = (d / divA) * mulA + (d % modB) * mulB;
  const float adv = ad[d];
  float e[10];
  float mx = -1e30f;
#pragma unroll
  for (int i = 0; i < 10; ++i) {
    float x = as[base + i * stride] + adv;
    x = (x > 0.0f) ? x : 0.2f * x;
    e[i] = x;
    mx = fmaxf(mx, x);
  }
  float ssum = 0.0f;
#pragma unroll
  for (int i = 0; i < 10; ++i) { e[i] = __expf(e[i] - mx); ssum += e[i]; }
  const float inv = 1.0f / ssum;
  for (int f = threadIdx.x; f < F; f += blockDim.x) {
    float acc = 0.0f;
#pragma unroll
    for (int i = 0; i < 10; ++i)
      acc += e[i] * zsrc[(size_t)(base + i * stride) * F + f];
    acc *= inv;
    out[(size_t)d * F + f] = (acc > 0.0f) ? acc : (__expf(acc) - 1.0f);
  }
}

// score += mean_r( sum_j tanh(T[r,j]) * q[j] ) — one wave per row, atomicAdd.
__global__ void sem_reduce(const float* __restrict__ T, const float* __restrict__ q,
                           float* __restrict__ score, int rows) {
  const int wv = (blockIdx.x * blockDim.x + threadIdx.x) >> 5;
  const int lane = threadIdx.x & 31;
  if (wv >= rows) return;
  const float* row = T + (size_t)wv * SEM_DIM;
  float s = 0.0f;
#pragma unroll
  for (int j = lane; j < SEM_DIM; j += 32) s += tanhf(row[j]) * q[j];
#pragma unroll
  for (int off = 16; off > 0; off >>= 1) s += __shfl_down(s, off, 32);
  if (lane == 0) atomicAdd(score, s / (float)rows);
}

// out = beta0*z0 + beta1*z1, beta = softmax(score[0], score[1])
__global__ void combine2(const float* __restrict__ z0, const float* __restrict__ z1,
                         const float* __restrict__ score, float* __restrict__ out,
                         unsigned long long n) {
  unsigned long long i = blockIdx.x * (unsigned long long)blockDim.x + threadIdx.x;
  if (i >= n) return;
  float s0 = score[0], s1 = score[1];
  float m = fmaxf(s0, s1);
  float e0 = __expf(s0 - m), e1 = __expf(s1 - m);
  float r = 1.0f / (e0 + e1);
  out[i] = (e0 * r) * z0[i] + (e1 * r) * z1[i];
}

__global__ void zero_scalars(float* __restrict__ p, int n) {
  if ((int)threadIdx.x < n) p[threadIdx.x] = 0.0f;
}

// ---------------------------------------------------------------------------
// Host-side orchestration
// ---------------------------------------------------------------------------
extern "C" void kernel_launch(void* const* d_in, const int* in_sizes, int n_in,
                              void* d_out, int out_size, void* d_ws, size_t ws_size,
                              hipStream_t stream) {
  (void)in_sizes; (void)n_in; (void)out_size; (void)ws_size;
  // Inputs in setup_inputs() dict order:
  const float* sentence = (const float*)d_in[0];
  const float* dtab     = (const float*)d_in[1];
  const float* Wp   = (const float*)d_in[2];  const float* bp   = (const float*)d_in[3];
  const float* W_ee = (const float*)d_in[4];  const float* al_ee= (const float*)d_in[5];  const float* ar_ee= (const float*)d_in[6];
  const float* W_pe = (const float*)d_in[7];  const float* al_pe= (const float*)d_in[8];  const float* ar_pe= (const float*)d_in[9];
  const float* W_cc = (const float*)d_in[10]; const float* al_cc= (const float*)d_in[11]; const float* ar_cc= (const float*)d_in[12];
  const float* W_pc = (const float*)d_in[13]; const float* al_pc= (const float*)d_in[14]; const float* ar_pc= (const float*)d_in[15];
  const float* Wq1  = (const float*)d_in[16]; const float* bq1  = (const float*)d_in[17]; const float* q1   = (const float*)d_in[18];
  const float* W_pep= (const float*)d_in[19]; const float* al_pep=(const float*)d_in[20]; const float* ar_pep=(const float*)d_in[21];
  const float* W_pcp= (const float*)d_in[22]; const float* al_pcp=(const float*)d_in[23]; const float* ar_pcp=(const float*)d_in[24];
  const float* Wq2  = (const float*)d_in[25]; const float* bq2  = (const float*)d_in[26]; const float* q2   = (const float*)d_in[27];
  const float* w_e  = (const float*)d_in[28]; const float* b_e  = (const float*)d_in[29];
  const float* w_c  = (const float*)d_in[30]; const float* b_c  = (const float*)d_in[31];
  const float* w_pp1= (const float*)d_in[32]; const float* b_pp1= (const float*)d_in[33];
  const float* W_pp3= (const float*)d_in[34]; const float* b_pp3= (const float*)d_in[35];
  const float* w_pp2= (const float*)d_in[36]; const float* b_pp2= (const float*)d_in[37];
  // d_in[38..49]: edge arrays (graphs are statically regular; unused)
  const int* emo_idx = (const int*)d_in[50];
  const int* cau_idx = (const int*)d_in[51];
  const int* ptag    = (const int*)d_in[52];

  float* out = (float*)d_out;   // [cpl_pred 6400 | cpl_pred2 6400 | pred_e 640 | pred_c 640]

  // ---- workspace carve-up (bytes, 256B aligned), sequential reuse ----
  char* wsb = (char*)d_ws;
  size_t o = 0;
  auto allocB = [&](size_t bytes) { char* p = wsb + o; o += (bytes + 255) & ~(size_t)255; return p; };
  auto allocF = [&](size_t n) { return (float*)allocB(n * sizeof(float)); };
  auto allocH = [&](size_t n) { return (unsigned short*)allocB(n * sizeof(unsigned short)); };

  float* scal  = allocF(8);                      // sem-attn scores (zeroed per call)
  float* pairB = allocF((size_t)NPAIR * D_DIM);  // pair, later h=relu(pair2@W_pp3)
  float* zbigA = allocF((size_t)NPAIR * D_DIM);  // z_pe_src, later z_pc_src
  float* zbigB = allocF((size_t)NPAIR * D2_DIM); // z_pep, z_pcp, later pair2
  float* pepB  = allocF((size_t)NPAIR * D2_DIM); // p_pep
  float* pcpB  = allocF((size_t)NPAIR * D2_DIM); // p_pcp
  float* zsent = allocF((size_t)NSENT * D_DIM);  // z_ee, later z_cc
  float* zsd   = allocF((size_t)NSENT * D_DIM);  // z_pe_dst, later z_pc_dst
  float* emoEE = allocF((size_t)NSENT * D_DIM);
  float* emoPE = allocF((size_t)NSENT * D_DIM);
  float* cauCC = allocF((size_t)NSENT * D_DIM);
  float* cauPC = allocF((size_t)NSENT * D_DIM);
  float* emoB  = allocF((size_t)NSENT * D_DIM);
  float* cauB  = allocF((size_t)NSENT * D_DIM);
  float* tbuf  = allocF((size_t)NPAIR * SEM_DIM);
  float* aS    = allocF(NPAIR);
  float* aD    = allocF(NPAIR);

  // bf16 split buffers
  unsigned short* ApadHi = allocH((size_t)NPAIR * DP_PAD);  // pair_raw / emo_cau split;
  unsigned short* ApadLo = allocH((size_t)NPAIR * DP_PAD);  //   overlaid as generic act split
  unsigned short* WtHi   = allocH((size_t)D2_DIM * DP_PAD); // transposed big weights
  unsigned short* WtLo   = allocH((size_t)D2_DIM * DP_PAD);
  unsigned short* WtQHi  = allocH((size_t)SEM_DIM * D2_DIM);// transposed Wq1/Wq2
  unsigned short* WtQLo  = allocH((size_t)SEM_DIM * D2_DIM);
  unsigned short* sentHi = allocH((size_t)NSENT * D_DIM);
  unsigned short* sentLo = allocH((size_t)NSENT * D_DIM);
  unsigned short* pairHi = allocH((size_t)NPAIR * D_DIM);
  unsigned short* pairLo = allocH((size_t)NPAIR * D_DIM);
  // generic activation split region (disjoint live range vs Apad usage)
  unsigned short* actHi = ApadHi;
  unsigned short* actLo = ApadLo;

  auto nblk = [](size_t n) { return (unsigned)((n + 255) / 256); };

  zero_scalars<<<1, 64, 0, stream>>>(scal, 8);

  // ---- sentence split (feeds 4 GEMMs) ----
  split_act<<<nblk((size_t)NSENT * D_DIM), 256, 0, stream>>>(sentence, sentHi, sentLo,
                                                             NSENT, D_DIM, D_DIM);

  // ---- pair = concat(sent[e], sent[c], dist) @ Wp + bp ----
  build_concat_split<<<NPAIR, 256, 0, stream>>>(sentence, sentence, dtab, emo_idx, cau_idx,
                                                ptag, ApadHi, ApadLo);
  split_weightT<<<nblk((size_t)D_DIM * DP_PAD), 256, 0, stream>>>(Wp, WtHi, WtLo,
                                                                  DP_DIM, D_DIM, DP_PAD);
  wmma_gemm_bf<<<dim3(NPAIR / 64, D_DIM / 64), 128, 0, stream>>>(ApadHi, ApadLo, WtHi, WtLo,
                                                                 bp, pairB, D_DIM, DP_PAD, 0);
  rowdot<<<NPAIR / 8, 256, 0, stream>>>(pairB, w_pp1, b_pp1, out + NPAIR, NPAIR, D_DIM);
  split_act<<<nblk((size_t)NPAIR * D_DIM), 256, 0, stream>>>(pairB, pairHi, pairLo,
                                                             NPAIR, D_DIM, D_DIM);

  // ---- GAT EE: base=(d/10)*10, stride 1 ----
  split_weightT<<<nblk((size_t)D_DIM * D_DIM), 256, 0, stream>>>(W_ee, WtHi, WtLo,
                                                                 D_DIM, D_DIM, D_DIM);
  wmma_gemm_bf<<<dim3(NSENT / 64, D_DIM / 64), 128, 0, stream>>>(sentHi, sentLo, WtHi, WtLo,
                                                                 nullptr, zsent, D_DIM, D_DIM, 0);
  rowdot<<<NSENT / 8, 256, 0, stream>>>(zsent, al_ee, nullptr, aS, NSENT, D_DIM);
  rowdot<<<NSENT / 8, 256, 0, stream>>>(zsent, ar_ee, nullptr, aD, NSENT, D_DIM);
  gat_agg<<<NSENT, 256, 0, stream>>>(zsent, aS, aD, emoEE, D_DIM, 10, 10, 10, 0, 1);

  // ---- GAT PE: base=(d/10)*100+(d%10)*10, stride 1 ----
  split_weightT<<<nblk((size_t)D_DIM * D_DIM), 256, 0, stream>>>(W_pe, WtHi, WtLo,
                                                                 D_DIM, D_DIM, D_DIM);
  wmma_gemm_bf<<<dim3(NPAIR / 64, D_DIM / 64), 128, 0, stream>>>(pairHi, pairLo, WtHi, WtLo,
                                                                 nullptr, zbigA, D_DIM, D_DIM, 0);
  wmma_gemm_bf<<<dim3(NSENT / 64, D_DIM / 64), 128, 0, stream>>>(sentHi, sentLo, WtHi, WtLo,
                                                                 nullptr, zsd, D_DIM, D_DIM, 0);
  rowdot<<<NPAIR / 8, 256, 0, stream>>>(zbigA, al_pe, nullptr, aS, NPAIR, D_DIM);
  rowdot<<<NSENT / 8, 256, 0, stream>>>(zsd, ar_pe, nullptr, aD, NSENT, D_DIM);
  gat_agg<<<NSENT, 256, 0, stream>>>(zbigA, aS, aD, emoPE, D_DIM, 10, 100, 10, 10, 1);

  // ---- GAT CC (EE topology) ----
  split_weightT<<<nblk((size_t)D_DIM * D_DIM), 256, 0, stream>>>(W_cc, WtHi, WtLo,
                                                                 D_DIM, D_DIM, D_DIM);
  wmma_gemm_bf<<<dim3(NSENT / 64, D_DIM / 64), 128, 0, stream>>>(sentHi, sentLo, WtHi, WtLo,
                                                                 nullptr, zsent, D_DIM, D_DIM, 0);
  rowdot<<<NSENT / 8, 256, 0, stream>>>(zsent, al_cc, nullptr, aS, NSENT, D_DIM);
  rowdot<<<NSENT / 8, 256, 0, stream>>>(zsent, ar_cc, nullptr, aD, NSENT, D_DIM);
  gat_agg<<<NSENT, 256, 0, stream>>>(zsent, aS, aD, cauCC, D_DIM, 10, 10, 10, 0, 1);

  // ---- GAT PC: base=(d/10)*100+(d%10), stride 10 ----
  split_weightT<<<nblk((size_t)D_DIM * D_DIM), 256, 0, stream>>>(W_pc, WtHi, WtLo,
                                                                 D_DIM, D_DIM, D_DIM);
  wmma_gemm_bf<<<dim3(NPAIR / 64, D_DIM / 64), 128, 0, stream>>>(pairHi, pairLo, WtHi, WtLo,
                                                                 nullptr, zbigA, D_DIM, D_DIM, 0);
  wmma_gemm_bf<<<dim3(NSENT / 64, D_DIM / 64), 128, 0, stream>>>(sentHi, sentLo, WtHi, WtLo,
                                                                 nullptr, zsd, D_DIM, D_DIM, 0);
  rowdot<<<NPAIR / 8, 256, 0, stream>>>(zbigA, al_pc, nullptr, aS, NPAIR, D_DIM);
  rowdot<<<NSENT / 8, 256, 0, stream>>>(zsd, ar_pc, nullptr, aD, NSENT, D_DIM);
  gat_agg<<<NSENT, 256, 0, stream>>>(zbigA, aS, aD, cauPC, D_DIM, 10, 100, 10, 1, 10);

  // ---- semantic attention level 1 (Wq1 transposed once; act region is free) ----
  split_weightT<<<nblk((size_t)SEM_DIM * D_DIM), 256, 0, stream>>>(Wq1, WtQHi, WtQLo,
                                                                   D_DIM, SEM_DIM, D_DIM);
  const float* semIn1[4] = { emoEE, emoPE, cauCC, cauPC };
  for (int i = 0; i < 4; ++i) {
    split_act<<<nblk((size_t)NSENT * D_DIM), 256, 0, stream>>>(semIn1[i], actHi, actLo,
                                                               NSENT, D_DIM, D_DIM);
    wmma_gemm_bf<<<dim3(NSENT / 64, SEM_DIM / 64), 128, 0, stream>>>(actHi, actLo, WtQHi, WtQLo,
                                                                     bq1, tbuf, SEM_DIM, D_DIM, 0);
    sem_reduce<<<NSENT / 8, 256, 0, stream>>>(tbuf, q1, scal + i, NSENT);
  }
  combine2<<<nblk((size_t)NSENT * D_DIM), 256, 0, stream>>>(emoEE, emoPE, scal + 0, emoB,
                                                            (unsigned long long)NSENT * D_DIM);
  combine2<<<nblk((size_t)NSENT * D_DIM), 256, 0, stream>>>(cauCC, cauPC, scal + 2, cauB,
                                                            (unsigned long long)NSENT * D_DIM);

  // ---- sentence heads ----
  rowdot<<<NSENT / 8, 256, 0, stream>>>(emoB, w_e, b_e, out + 2 * NPAIR, NSENT, D_DIM);
  rowdot<<<NSENT / 8, 256, 0, stream>>>(cauB, w_c, b_c, out + 2 * NPAIR + NSENT, NSENT, D_DIM);

  // ---- emo_cau = concat(emo[e], cau[c], dist) -> split (reclaims act region) ----
  build_concat_split<<<NPAIR, 256, 0, stream>>>(emoB, cauB, dtab, emo_idx, cau_idx, ptag,
                                                ApadHi, ApadLo);

  // ---- GAT PEP: base=(d/10)*10, stride 1 ----
  split_weightT<<<nblk((size_t)D2_DIM * DP_PAD), 256, 0, stream>>>(W_pep, WtHi, WtLo,
                                                                   DP_DIM, D2_DIM, DP_PAD);
  wmma_gemm_bf<<<dim3(NPAIR / 64, D2_DIM / 64), 128, 0, stream>>>(ApadHi, ApadLo, WtHi, WtLo,
                                                                  nullptr, zbigB, D2_DIM, DP_PAD, 0);
  rowdot<<<NPAIR / 8, 256, 0, stream>>>(zbigB, al_pep, nullptr, aS, NPAIR, D2_DIM);
  rowdot<<<NPAIR / 8, 256, 0, stream>>>(zbigB, ar_pep, nullptr, aD, NPAIR, D2_DIM);
  gat_agg<<<NPAIR, 256, 0, stream>>>(zbigB, aS, aD, pepB, D2_DIM, 10, 10, 10, 0, 1);

  // ---- GAT PCP: base=(d/100)*100+(d%10), stride 10 ----
  split_weightT<<<nblk((size_t)D2_DIM * DP_PAD), 256, 0, stream>>>(W_pcp, WtHi, WtLo,
                                                                   DP_DIM, D2_DIM, DP_PAD);
  wmma_gemm_bf<<<dim3(NPAIR / 64, D2_DIM / 64), 128, 0, stream>>>(ApadHi, ApadLo, WtHi, WtLo,
                                                                  nullptr, zbigB, D2_DIM, DP_PAD, 0);
  rowdot<<<NPAIR / 8, 256, 0, stream>>>(zbigB, al_pcp, nullptr, aS, NPAIR, D2_DIM);
  rowdot<<<NPAIR / 8, 256, 0, stream>>>(zbigB, ar_pcp, nullptr, aD, NPAIR, D2_DIM);
  gat_agg<<<NPAIR, 256, 0, stream>>>(zbigB, aS, aD, pcpB, D2_DIM, 100, 100, 10, 1, 10);

  // ---- semantic attention level 2 (Apad now dead -> act region free) ----
  split_weightT<<<nblk((size_t)SEM_DIM * D2_DIM), 256, 0, stream>>>(Wq2, WtQHi, WtQLo,
                                                                    D2_DIM, SEM_DIM, D2_DIM);
  const float* semIn2[2] = { pepB, pcpB };
  for (int i = 0; i < 2; ++i) {
    split_act<<<nblk((size_t)NPAIR * D2_DIM), 256, 0, stream>>>(semIn2[i], actHi, actLo,
                                                                NPAIR, D2_DIM, D2_DIM);
    wmma_gemm_bf<<<dim3(NPAIR / 64, SEM_DIM / 64), 128, 0, stream>>>(actHi, actLo, WtQHi, WtQLo,
                                                                     bq2, tbuf, SEM_DIM, D2_DIM, 0);
    sem_reduce<<<NPAIR / 8, 256, 0, stream>>>(tbuf, q2, scal + 4 + i, NPAIR);
  }
  combine2<<<nblk((size_t)NPAIR * D2_DIM), 256, 0, stream>>>(pepB, pcpB, scal + 4, zbigB,
                                                             (unsigned long long)NPAIR * D2_DIM);

  // ---- final MLP: h = relu(pair2 @ W_pp3 + b_pp3); couples_pred = h@w_pp2+b ----
  split_act<<<nblk((size_t)NPAIR * D2_DIM), 256, 0, stream>>>(zbigB, actHi, actLo,
                                                              NPAIR, D2_DIM, D2_DIM);
  split_weightT<<<nblk((size_t)D_DIM * D2_DIM), 256, 0, stream>>>(W_pp3, WtHi, WtLo,
                                                                  D2_DIM, D_DIM, D2_DIM);
  wmma_gemm_bf<<<dim3(NPAIR / 64, D_DIM / 64), 128, 0, stream>>>(actHi, actLo, WtHi, WtLo,
                                                                 b_pp3, pairB, D_DIM, D2_DIM, 1);
  rowdot<<<NPAIR / 8, 256, 0, stream>>>(pairB, w_pp2, b_pp2, out, NPAIR, D_DIM);
}
